// RNNAttention_23545010717391
// MI455X (gfx1250) — matrix operand order repaired and checked by reading, not compiled, not run on hardware
//
#include <hip/hip_runtime.h>

#define B_ 4
#define T_ 2048
#define E_ 256
#define H_ 512
#define V_ 32000
#define BT_ (B_*T_)                 // 8192
#define LOGITS_ ((long)BT_*V_)      // 262,144,000

typedef __attribute__((ext_vector_type(16))) __bf16 v16bf;
typedef __attribute__((ext_vector_type(8)))  float v8f;

__device__ __forceinline__ v8f wmma_bf16(v16bf a, v16bf b, v8f c){
  return __builtin_amdgcn_wmma_f32_16x16x32_bf16(false, a, false, b, (short)0, c, false, false);
}

// A-matrix 16x32 bf16 fragment, row contiguous in memory (K along row).
// Per ISA: lane half h holds K in [8h,8h+8) and [16+8h,16+8h+8).
__device__ __forceinline__ v16bf loadA_bf16(const __bf16* rowk, int half){
  union { uint4 u[2]; v16bf h; } f;
  f.u[0] = *(const uint4*)(rowk + half*8);
  f.u[1] = *(const uint4*)(rowk + 16 + half*8);
  return f.h;
}
// B-matrix 32x16 bf16 fragment; lane holds column n with K in [8h,8h+8) and [16+8h,16+8h+8).
__device__ __forceinline__ v16bf loadB_bf16(const __bf16* colk, int half){
  union { uint4 u[2]; v16bf h; } f;
  f.u[0] = *(const uint4*)(colk + half*8);
  f.u[1] = *(const uint4*)(colk + 16 + half*8);
  return f.h;
}
// A fragment built from an f32 row (native cvt on the fly).
__device__ __forceinline__ v16bf loadA_f32(const float* rowk, int half){
  const float4* p0 = (const float4*)(rowk + half*8);
  const float4* p1 = (const float4*)(rowk + 16 + half*8);
  float4 a = p0[0], b = p0[1], c = p1[0], d = p1[1];
  float v[16] = {a.x,a.y,a.z,a.w,b.x,b.y,b.z,b.w,c.x,c.y,c.z,c.w,d.x,d.y,d.z,d.w};
  union { __bf16 s[16]; v16bf h; } f;
#pragma unroll
  for (int i=0;i<16;i++) f.s[i] = (__bf16)v[i];
  return f.h;
}

// ---------------- kernel 0: f32 -> bf16 weight conversion ----------------
__global__ void k_convert(const float* w_ih, const float* w_hh, const float* fc_w,
                          __bf16* w_ih_b, __bf16* w_hh_b, __bf16* fc_w_b){
  long i0 = (long)blockIdx.x*blockDim.x + threadIdx.x;
  long st = (long)gridDim.x*blockDim.x;
  for (long i=i0;i<(long)H_*E_;  i+=st) w_ih_b[i] = (__bf16)w_ih[i];
  for (long i=i0;i<(long)H_*H_;  i+=st) w_hh_b[i] = (__bf16)w_hh[i];
  for (long i=i0;i<(long)V_*2*H_;i+=st) fc_w_b[i] = (__bf16)fc_w[i];
}

// ---------------- kernel 1: embedding gather + input projection ----------------
// xp[8192,512] = embed_w[x] @ w_ih^T + b_ih ; one 16x16 tile per wave.
__global__ void __launch_bounds__(256) k_embed_proj(const int* x, const float* embed_w,
                                                    const __bf16* w_ih_b,
                                                    const float* b_ih, float* xp){
  int wid  = blockIdx.x*8 + (threadIdx.x>>5);   // 16384 wave-tiles
  int lane = threadIdx.x & 31;
  int mt = wid >> 5, nt = wid & 31;             // 512 x 32 tiles
  int m = lane & 15, half = lane >> 4;
  int tok = x[mt*16 + m];
  const float*  arow = embed_w + (long)tok * E_;
  const __bf16* brow = w_ih_b + (long)(nt*16 + m) * E_;
  v8f c = {0.f,0.f,0.f,0.f,0.f,0.f,0.f,0.f};
#pragma unroll
  for (int kk=0; kk<E_; kk+=32)
    c = wmma_bf16(loadA_f32(arow + kk, half), loadB_bf16(brow + kk, half), c);
  int col = nt*16 + m;
  float bias = b_ih[col];
#pragma unroll
  for (int r=0;r<8;r++)
    xp[(long)(mt*16 + r + 8*half)*H_ + col] = c[r] + bias;
}

// ---------------- kernel 2: sequential RNN scan (single WGP) ----------------
// 32 waves; wave w owns output cols [16w,16w+16). h kept in LDS as bf16 rows 0..15
// (rows 4..15 zero padding). 16 WMMA k-steps per step, two barriers per step.
__global__ void __launch_bounds__(1024) k_rnn(const float* xp, const __bf16* w_hh_b,
                                              const float* b_hh, const float* h0,
                                              __bf16* combined, float* out){
  __shared__ __align__(16) __bf16 hbuf[16*H_];
  int tid = threadIdx.x, wave = tid >> 5, lane = tid & 31;
  int m = lane & 15, half = lane >> 4;
  int col = wave*16 + m;
  for (int i = tid; i < 16*H_; i += blockDim.x){
    int r = i / H_, cc = i - r*H_;
    hbuf[i] = (r < B_) ? (__bf16)h0[r*H_ + cc] : (__bf16)0.f;
  }
  float bias = b_hh[col];
  const __bf16* brow = w_hh_b + (long)col * H_;
  const __bf16* arow = hbuf + m*H_;
  float last[4] = {0.f,0.f,0.f,0.f};
  for (int t=0; t<T_; t++){
    __syncthreads();                       // h writes from previous step visible
    v8f c = {0.f,0.f,0.f,0.f,0.f,0.f,0.f,0.f};
#pragma unroll
    for (int kk=0; kk<H_; kk+=32)
      c = wmma_bf16(loadA_bf16(arow + kk, half), loadB_bf16(brow + kk, half), c);
    __syncthreads();                       // all reads of h done before overwrite
    if (half == 0){
#pragma unroll
      for (int r=0;r<4;r++){               // rows M=r = batch b (only 0..3 valid)
        float v = tanhf(c[r] + xp[((long)r*T_ + t)*H_ + col] + bias);
        hbuf[r*H_ + col] = (__bf16)v;
        combined[((long)r*T_ + t)*(2*H_) + col] = (__bf16)v;
        last[r] = v;
      }
    }
  }
  if (half == 0){
#pragma unroll
    for (int r=0;r<4;r++) out[LOGITS_ + (long)r*H_ + col] = last[r];
  }
}

// ---------------- kernel 3: causal self-attention, one (b, query-tile) per block ----------------
// Full score row-block (16 x 2048 f32 = 128 KB) lives in the 320 KB WGP LDS. 3 phases.
__global__ void __launch_bounds__(256) k_attn(__bf16* combined){
  __shared__ float S[16*T_];
  __shared__ float red[256];
  __shared__ float rowmax[16], rowsum[16];
  int b  = blockIdx.x >> 7;
  int qt = blockIdx.x & 127;
  int tid = threadIdx.x, wave = tid>>5, lane = tid&31;
  int m = lane & 15, half = lane>>4;
  long base = (long)b * T_ * (2*H_);
  const __bf16* qrow = combined + base + (long)(qt*16 + m)*(2*H_);
  // phase 1: S = Q @ K^T with causal mask
  for (int kt = wave; kt <= qt; kt += 8){
    const __bf16* krow = combined + base + (long)(kt*16 + m)*(2*H_);
    v8f c = {0.f,0.f,0.f,0.f,0.f,0.f,0.f,0.f};
#pragma unroll
    for (int kk=0;kk<H_;kk+=32)
      c = wmma_bf16(loadA_bf16(qrow + kk, half), loadB_bf16(krow + kk, half), c);
#pragma unroll
    for (int r=0;r<8;r++){
      int mm = r + 8*half;
      float v = c[r];
      if (kt==qt && m > mm) v = -3.0e38f;
      S[mm*T_ + kt*16 + m] = v;
    }
  }
  __syncthreads();
  // phase 2: row softmax (store exp, keep row sums for final scale)
  int nk = (qt+1)*16;
  int row = tid>>4, sub = tid&15;
  float mx = -3.0e38f;
  for (int j=sub;j<nk;j+=16) mx = fmaxf(mx, S[row*T_+j]);
  red[tid]=mx; __syncthreads();
  if (sub==0){ float v=red[row*16]; for(int i=1;i<16;i++) v=fmaxf(v,red[row*16+i]); rowmax[row]=v; }
  __syncthreads();
  float rm = rowmax[row], sm = 0.f;
  for (int j=sub;j<nk;j+=16){ float e=__expf(S[row*T_+j]-rm); S[row*T_+j]=e; sm+=e; }
  red[tid]=sm; __syncthreads();
  if (sub==0){ float v=0.f; for(int i=0;i<16;i++) v+=red[row*16+i]; rowsum[row]=v; }
  __syncthreads();
  // phase 3: context = P @ V ; each wave owns 64 output cols
  int cbase = wave*64;
  v8f acc[4] = {};
  const __bf16* vbase = combined + base;
  int nk32 = (nk+31) & ~31;
  for (int kk=0;kk<nk32;kk+=32){
    union { __bf16 s[16]; v16bf h; } fa;
#pragma unroll
    for (int i=0;i<16;i++){
      int K = kk + half*8 + i + ((i>=8)?8:0);
      fa.s[i] = (__bf16)((K<nk) ? S[m*T_ + K] : 0.f);
    }
#pragma unroll
    for (int nt=0;nt<4;nt++){
      int n = cbase + nt*16 + m;
      union { __bf16 s[16]; v16bf h; } fb;
#pragma unroll
      for (int i=0;i<16;i++){
        int K = kk + half*8 + i + ((i>=8)?8:0);
        fb.s[i] = vbase[(long)K*(2*H_) + n];
      }
      acc[nt] = wmma_bf16(fa.h, fb.h, acc[nt]);
    }
  }
#pragma unroll
  for (int nt=0;nt<4;nt++){
#pragma unroll
    for (int r=0;r<8;r++){
      int mm = r + 8*half;
      float v = acc[nt][r] / rowsum[mm];
      combined[base + (long)(qt*16 + mm)*(2*H_) + H_ + cbase + nt*16 + m] = (__bf16)v;
    }
  }
}

// ---------------- kernel 4: output projection (dominant GEMM) ----------------
// logits[8192,32000] = combined @ fc_w^T + fc_b. Wave: 16 rows x 64 cols, A reused 4x.
__global__ void __launch_bounds__(128) k_fc(const __bf16* combined,
                                            const __bf16* fc_w_b,
                                            const float* fc_b, float* out){
  int wave = threadIdx.x >> 5, lane = threadIdx.x & 31;
  int m = lane & 15, half = lane >> 4;
  int chunk = blockIdx.x*4 + wave;          // 0..499 (64-col chunks)
  int mbase = blockIdx.y*16;
  const __bf16* arow  = combined + (long)(mbase + m)*(2*H_);
  const __bf16* brow0 = fc_w_b  + (long)(chunk*64 + m)*(2*H_);
  v8f acc[4] = {};
#pragma unroll 4
  for (int kk=0; kk<2*H_; kk+=32){
    if (kk + 32 < 2*H_) __builtin_prefetch(arow + kk + 32, 0, 3);  // global_prefetch next A chunk
    v16bf a = loadA_bf16(arow + kk, half);
#pragma unroll
    for (int nt=0;nt<4;nt++)
      acc[nt] = wmma_bf16(a, loadB_bf16(brow0 + (long)nt*16*(2*H_) + kk, half), acc[nt]);
  }
#pragma unroll
  for (int nt=0;nt<4;nt++){
    int col = chunk*64 + nt*16 + m;
    float bias = fc_b[col];
#pragma unroll
    for (int r=0;r<8;r++)
      out[(long)(mbase + r + 8*half)*V_ + col] = acc[nt][r] + bias;
  }
}

extern "C" void kernel_launch(void* const* d_in, const int* in_sizes, int n_in,
                              void* d_out, int out_size, void* d_ws, size_t ws_size,
                              hipStream_t stream) {
  (void)in_sizes; (void)n_in; (void)out_size; (void)ws_size;
  const int*   x       = (const int*)  d_in[0];
  const float* embed_w = (const float*)d_in[1];
  const float* w_ih    = (const float*)d_in[2];
  const float* w_hh    = (const float*)d_in[3];
  const float* b_ih    = (const float*)d_in[4];
  const float* b_hh    = (const float*)d_in[5];
  const float* fc_w    = (const float*)d_in[6];
  const float* fc_b    = (const float*)d_in[7];
  const float* h0      = (const float*)d_in[8];
  float* out = (float*)d_out;
  char*  ws  = (char*)d_ws;

  float*  xp       = (float*)ws;                            // 16,777,216 B
  __bf16* combined = (__bf16*)(ws + (size_t)16777216);      // 16,777,216 B
  __bf16* w_ih_b   = (__bf16*)(ws + (size_t)33554432);      //    262,144 B
  __bf16* w_hh_b   = (__bf16*)(ws + (size_t)33816576);      //    524,288 B
  __bf16* fc_w_b   = (__bf16*)(ws + (size_t)34340864);      // 65,536,000 B

  k_convert   <<<2048, 256, 0, stream>>>(w_ih, w_hh, fc_w, w_ih_b, w_hh_b, fc_w_b);
  k_embed_proj<<<2048, 256, 0, stream>>>(x, embed_w, w_ih_b, b_ih, xp);
  k_rnn       <<<1, 1024, 0, stream>>>(xp, w_hh_b, b_hh, h0, combined, out);
  k_attn      <<<512, 256, 0, stream>>>(combined);
  k_fc        <<<dim3(125, 512), 128, 0, stream>>>(combined, fc_w_b, fc_b, out);
}
